// MultiHeadAttention_58213986730143
// MI455X (gfx1250) — matrix operand order repaired and checked by reading, not compiled, not run on hardware
//
#include <hip/hip_runtime.h>
#include <hip/hip_bf16.h>

// ---------------------------------------------------------------------------
// Multi-head attention for MI455X (gfx1250), bf16 WMMA path.
//   EMBED=1024, H=16, D=64, S=2048, B=2  -> M = B*S = 4096 token rows.
// ~69 GFLOP total vs a few hundred MB of traffic -> compute bound; all
// matmuls go through v_wmma_f32_16x16x32_bf16 (fp32 accumulate). Weights are
// L2-resident (2MB bf16 vs 192MB L2); GEMM waves compute 32x64 tiles to
// maximize WMMA-per-byte. Score scale 1/sqrt(64) folded into Q projection.
// ---------------------------------------------------------------------------

#define EMBED 1024
#define NHEAD 16
#define HDIM  64
#define SEQ   2048
#define BATCH 2
#define MROWS (BATCH * SEQ) /* 4096 */

typedef __attribute__((ext_vector_type(16))) __bf16 v16bf;
typedef __attribute__((ext_vector_type(8)))  float  v8f;

#define V8F_ZERO {0.f,0.f,0.f,0.f,0.f,0.f,0.f,0.f}

__device__ __forceinline__ unsigned short f2bf(float f) {
  unsigned u = __builtin_bit_cast(unsigned, f);
  u += 0x7FFFu + ((u >> 16) & 1u);            // round-to-nearest-even
  return (unsigned short)(u >> 16);
}

// Build a 16-element bf16 fragment from two contiguous 8-element (16B) chunks.
__device__ __forceinline__ v16bf frag_ld(const unsigned short* p0,
                                         const unsigned short* p1) {
  union { uint4 u[2]; v16bf v; } t;
  t.u[0] = *reinterpret_cast<const uint4*>(p0);
  t.u[1] = *reinterpret_cast<const uint4*>(p1);
  return t.v;
}

__device__ __forceinline__ v8f wmma_bf16(v16bf a, v16bf b, v8f c) {
  // (neg_a, A, neg_b, B, c_mod, C, reuse_a, reuse_b)
  return __builtin_amdgcn_wmma_f32_16x16x32_bf16(false, a, false, b,
                                                 (short)0, c, false, false);
}

// ---------------------------------------------------------------------------
__global__ void cvt_bf16_kernel(const float* __restrict__ in,
                                unsigned short* __restrict__ out, int n) {
  int i = blockIdx.x * blockDim.x + threadIdx.x;
  int stride = gridDim.x * blockDim.x;
  for (; i < n; i += stride) out[i] = f2bf(in[i]);
}

// ---------------------------------------------------------------------------
// Y = (X @ W^T + bias) * outScale, X:[MROWS,EMBED] bf16, W:[EMBED,EMBED] bf16
// (row = output feature).  Output bf16 in head-split layout:
//   (row,col) -> (b,s,h,d) at (b*NHEAD+h)*SEQ*HDIM + s*sStride + d*dStride
// (sStride=64,dStride=1 -> [B,H,S,D];  sStride=1,dStride=SEQ -> [B,H,D,S]).
// Block: 256 thr = 8 waves (4 M-groups x 2 N-groups); wave tile 32x64;
// block tile 128x128.  8 WMMAs per k-step per wave, B fragments reused 2x.
__global__ void gemm_proj_bf16(const unsigned short* __restrict__ A,
                               const unsigned short* __restrict__ W,
                               const float* __restrict__ bias,
                               unsigned short* __restrict__ Y,
                               int sStride, int dStride, float outScale) {
  const int lane = threadIdx.x & 31, wave = threadIdx.x >> 5;
  const int half = lane >> 4, nn = lane & 15;
  const int mtile = blockIdx.x * 128 + (wave & 3) * 32;
  const int nbase = blockIdx.y * 128 + (wave >> 2) * 64;
  const unsigned short* aptr0 = A + (size_t)(mtile + nn) * EMBED;
  const unsigned short* aptr1 = A + (size_t)(mtile + 16 + nn) * EMBED;

  v8f acc[2][4] = {{V8F_ZERO, V8F_ZERO, V8F_ZERO, V8F_ZERO},
                   {V8F_ZERO, V8F_ZERO, V8F_ZERO, V8F_ZERO}};

  for (int kk = 0; kk < EMBED; kk += 32) {
    v16bf af0 = frag_ld(aptr0 + kk + half * 16, aptr0 + kk + half * 16 + 8);
    v16bf af1 = frag_ld(aptr1 + kk + half * 16, aptr1 + kk + half * 16 + 8);
#pragma unroll
    for (int nt = 0; nt < 4; ++nt) {
      const int col = nbase + nt * 16 + nn;
      const unsigned short* wp = W + (size_t)col * EMBED + kk + half * 16;
      v16bf bf = frag_ld(wp, wp + 8);  // B[kk..kk+31][col], 16 contiguous k
      acc[0][nt] = wmma_bf16(af0, bf, acc[0][nt]);
      acc[1][nt] = wmma_bf16(af1, bf, acc[1][nt]);
    }
  }

#pragma unroll
  for (int mg = 0; mg < 2; ++mg) {
#pragma unroll
    for (int nt = 0; nt < 4; ++nt) {
      const int col = nbase + nt * 16 + nn;
      const int h = col >> 6, d = col & 63;
      const float bv = bias[col];
#pragma unroll
      for (int v = 0; v < 8; ++v) {
        const int row = mtile + mg * 16 + v + 8 * half;
        const int bb = row >> 11, s = row & (SEQ - 1);
        size_t idx = (size_t)(bb * NHEAD + h) * (size_t)(SEQ * HDIM) +
                     (size_t)s * (size_t)sStride + (size_t)d * (size_t)dStride;
        Y[idx] = f2bf((acc[mg][nt][v] + bv) * outScale);
      }
    }
  }
}

// Same 32x64-per-wave GEMM but fp32 row-major output (final projection).
__global__ void gemm_out_f32(const unsigned short* __restrict__ A,
                             const unsigned short* __restrict__ W,
                             const float* __restrict__ bias,
                             float* __restrict__ Y) {
  const int lane = threadIdx.x & 31, wave = threadIdx.x >> 5;
  const int half = lane >> 4, nn = lane & 15;
  const int mtile = blockIdx.x * 128 + (wave & 3) * 32;
  const int nbase = blockIdx.y * 128 + (wave >> 2) * 64;
  const unsigned short* aptr0 = A + (size_t)(mtile + nn) * EMBED;
  const unsigned short* aptr1 = A + (size_t)(mtile + 16 + nn) * EMBED;

  v8f acc[2][4] = {{V8F_ZERO, V8F_ZERO, V8F_ZERO, V8F_ZERO},
                   {V8F_ZERO, V8F_ZERO, V8F_ZERO, V8F_ZERO}};

  for (int kk = 0; kk < EMBED; kk += 32) {
    v16bf af0 = frag_ld(aptr0 + kk + half * 16, aptr0 + kk + half * 16 + 8);
    v16bf af1 = frag_ld(aptr1 + kk + half * 16, aptr1 + kk + half * 16 + 8);
#pragma unroll
    for (int nt = 0; nt < 4; ++nt) {
      const int col = nbase + nt * 16 + nn;
      const unsigned short* wp = W + (size_t)col * EMBED + kk + half * 16;
      v16bf bf = frag_ld(wp, wp + 8);
      acc[0][nt] = wmma_bf16(af0, bf, acc[0][nt]);
      acc[1][nt] = wmma_bf16(af1, bf, acc[1][nt]);
    }
  }

#pragma unroll
  for (int mg = 0; mg < 2; ++mg) {
#pragma unroll
    for (int nt = 0; nt < 4; ++nt) {
      const int col = nbase + nt * 16 + nn;
      const float bv = bias[col];
#pragma unroll
      for (int v = 0; v < 8; ++v) {
        const int row = mtile + mg * 16 + v + 8 * half;
        Y[(size_t)row * EMBED + col] = acc[mg][nt][v] + bv;
      }
    }
  }
}

// ---------------------------------------------------------------------------
// Flash-style attention.  Grid: (SEQ/128, NHEAD, BATCH), block 256 = 8 waves.
// Each wave owns a 16-row Q tile, streams keys in chunks of 32.
// Qp (pre-scaled by 1/sqrt(64)), Kp: [B,H,S,64] bf16.  VT: [B,H,64,S] bf16.
// O: [B,S,EMBED] bf16 (row-major attn output, feeds final GEMM).
__global__ void attn_kernel(const unsigned short* __restrict__ Qp,
                            const unsigned short* __restrict__ Kp,
                            const unsigned short* __restrict__ VT,
                            unsigned short* __restrict__ O) {
  __shared__ __align__(16) unsigned short pbuf[8][16][32]; // 8KB: P tiles

  const int lane = threadIdx.x & 31, wave = threadIdx.x >> 5;
  const int half = lane >> 4, nn = lane & 15;
  const int b = blockIdx.z, h = blockIdx.y;
  const int qtile = blockIdx.x * 128 + wave * 16;

  const size_t headoff = (size_t)(b * NHEAD + h) * (size_t)(SEQ * HDIM);
  const unsigned short* Qh = Qp + headoff;
  const unsigned short* Kh = Kp + headoff;
  const unsigned short* Vh = VT + headoff;

  // Q A-fragments for d-chunks [0,32) and [32,64)
  const unsigned short* qrow = Qh + (size_t)(qtile + nn) * HDIM;
  const v16bf qa0 = frag_ld(qrow + half * 16, qrow + half * 16 + 8);
  const v16bf qa1 = frag_ld(qrow + 32 + half * 16, qrow + 32 + half * 16 + 8);

  v8f acc[4] = {V8F_ZERO, V8F_ZERO, V8F_ZERO, V8F_ZERO};
  float mrow[8], lrow[8];
#pragma unroll
  for (int v = 0; v < 8; ++v) { mrow[v] = -1.0e30f; lrow[v] = 0.0f; }

  for (int kc = 0; kc < SEQ; kc += 32) {
    // ---- scores S = Q . K^T for 32 keys (two 16-key n-tiles) ----
    v8f s0 = V8F_ZERO;
    v8f s1 = V8F_ZERO;
    {
      const unsigned short* kr0 = Kh + (size_t)(kc + nn) * HDIM + half * 16;
      const unsigned short* kr1 = Kh + (size_t)(kc + 16 + nn) * HDIM + half * 16;
      v16bf kb00 = frag_ld(kr0, kr0 + 8);        // d in [0,32)
      v16bf kb01 = frag_ld(kr0 + 32, kr0 + 40);  // d in [32,64)
      v16bf kb10 = frag_ld(kr1, kr1 + 8);
      v16bf kb11 = frag_ld(kr1 + 32, kr1 + 40);
      s0 = wmma_bf16(qa0, kb00, s0);
      s0 = wmma_bf16(qa1, kb01, s0);
      s1 = wmma_bf16(qa0, kb10, s1);
      s1 = wmma_bf16(qa1, kb11, s1);
    }

    // ---- online softmax; row m = v + 8*half lives across the 16-lane half
    float sc[8], p0[8], p1[8];
#pragma unroll
    for (int v = 0; v < 8; ++v) {
      const float a = s0[v];   // Q pre-scaled by 1/sqrt(HDIM)
      const float c = s1[v];
      float t = fmaxf(a, c);
#pragma unroll
      for (int msk = 1; msk < 16; msk <<= 1)
        t = fmaxf(t, __shfl_xor(t, msk, 32));
      const float mn = fmaxf(mrow[v], t);
      const float scv = __expf(mrow[v] - mn);
      mrow[v] = mn;
      sc[v] = scv;
      const float e0 = __expf(a - mn);
      const float e1 = __expf(c - mn);
      p0[v] = e0; p1[v] = e1;
      float rs = e0 + e1;
#pragma unroll
      for (int msk = 1; msk < 16; msk <<= 1) rs += __shfl_xor(rs, msk, 32);
      lrow[v] = lrow[v] * scv + rs;
    }
#pragma unroll
    for (int dt = 0; dt < 4; ++dt)
#pragma unroll
      for (int v = 0; v < 8; ++v) acc[dt][v] *= sc[v];

    // ---- relayout P: C-fragment -> A-fragment via LDS (bf16) ----
#pragma unroll
    for (int v = 0; v < 8; ++v) {
      const int row = v + 8 * half;
      pbuf[wave][row][nn]      = f2bf(p0[v]);
      pbuf[wave][row][16 + nn] = f2bf(p1[v]);
    }
    asm volatile("s_wait_dscnt 0" ::: "memory");
    const v16bf pa = frag_ld(&pbuf[wave][nn][half * 16],
                             &pbuf[wave][nn][half * 16 + 8]);
    asm volatile("s_wait_dscnt 0" ::: "memory");

    // ---- O += P @ V  (V^T rows contiguous in key dimension) ----
#pragma unroll
    for (int dt = 0; dt < 4; ++dt) {
      const unsigned short* vp =
          Vh + (size_t)(dt * 16 + nn) * SEQ + kc + half * 16;
      v16bf vb = frag_ld(vp, vp + 8);
      acc[dt] = wmma_bf16(pa, vb, acc[dt]);
    }
  }

  // ---- epilogue: divide by l, store attn in [B,S,EMBED] bf16 ----
#pragma unroll
  for (int dt = 0; dt < 4; ++dt) {
#pragma unroll
    for (int v = 0; v < 8; ++v) {
      const int row = v + 8 * half;
      const int q = qtile + row;
      const float val = acc[dt][v] / lrow[v];
      O[((size_t)b * SEQ + q) * EMBED + h * HDIM + dt * 16 + nn] = f2bf(val);
    }
  }
}

// ---------------------------------------------------------------------------
extern "C" void kernel_launch(void* const* d_in, const int* in_sizes, int n_in,
                              void* d_out, int out_size, void* d_ws,
                              size_t ws_size, hipStream_t stream) {
  const float* Qin = (const float*)d_in[0];
  const float* Kin = (const float*)d_in[1];
  const float* Vin = (const float*)d_in[2];
  const float* Wq  = (const float*)d_in[3];
  const float* bq  = (const float*)d_in[4];
  const float* Wk  = (const float*)d_in[5];
  const float* bk  = (const float*)d_in[6];
  const float* Wv  = (const float*)d_in[7];
  const float* bv  = (const float*)d_in[8];
  const float* Wo  = (const float*)d_in[9];
  const float* bo  = (const float*)d_in[10];
  float* out = (float*)d_out;

  const size_t NE_IN = (size_t)MROWS * EMBED;  // 4 Mi elements
  const size_t NE_W  = (size_t)EMBED * EMBED;  // 1 Mi elements

  unsigned short* p = (unsigned short*)d_ws;
  unsigned short* qin_b = p; p += NE_IN;
  unsigned short* kin_b = p; p += NE_IN;
  unsigned short* vin_b = p; p += NE_IN;
  unsigned short* wq_b  = p; p += NE_W;
  unsigned short* wk_b  = p; p += NE_W;
  unsigned short* wv_b  = p; p += NE_W;
  unsigned short* wo_b  = p; p += NE_W;
  unsigned short* Qp    = p; p += NE_IN;  // [B,H,S,64], pre-scaled by 1/8
  unsigned short* Kp    = p; p += NE_IN;  // [B,H,S,64]
  unsigned short* VT    = p; p += NE_IN;  // [B,H,64,S]
  unsigned short* attnb = p; p += NE_IN;  // [B,S,EMBED]

  // 1) fp32 -> bf16 conversions
  cvt_bf16_kernel<<<1024, 256, 0, stream>>>(Qin, qin_b, (int)NE_IN);
  cvt_bf16_kernel<<<1024, 256, 0, stream>>>(Kin, kin_b, (int)NE_IN);
  cvt_bf16_kernel<<<1024, 256, 0, stream>>>(Vin, vin_b, (int)NE_IN);
  cvt_bf16_kernel<<<1024, 256, 0, stream>>>(Wq, wq_b, (int)NE_W);
  cvt_bf16_kernel<<<1024, 256, 0, stream>>>(Wk, wk_b, (int)NE_W);
  cvt_bf16_kernel<<<1024, 256, 0, stream>>>(Wv, wv_b, (int)NE_W);
  cvt_bf16_kernel<<<1024, 256, 0, stream>>>(Wo, wo_b, (int)NE_W);

  // 2) Q/K/V projections (WMMA); V stored transposed per head.
  //    Q pre-scaled by 1/sqrt(HDIM) = 0.125 (exact power of two).
  dim3 gg(MROWS / 128, EMBED / 128), bb256(256);
  gemm_proj_bf16<<<gg, bb256, 0, stream>>>(qin_b, wq_b, bq, Qp, HDIM, 1,
                                           0.125f);
  gemm_proj_bf16<<<gg, bb256, 0, stream>>>(kin_b, wk_b, bk, Kp, HDIM, 1, 1.0f);
  gemm_proj_bf16<<<gg, bb256, 0, stream>>>(vin_b, wv_b, bv, VT, 1, SEQ, 1.0f);

  // 3) flash attention (WMMA for QK^T and P.V)
  attn_kernel<<<dim3(SEQ / 128, NHEAD, BATCH), 256, 0, stream>>>(Qp, Kp, VT,
                                                                 attnb);

  // 4) output projection, fp32 result
  gemm_out_f32<<<gg, bb256, 0, stream>>>(attnb, wo_b, bo, out);
}
